// Orient_23356032156287
// MI455X (gfx1250) — matrix operand order repaired and checked by reading, not compiled, not run on hardware
//
#include <hip/hip_runtime.h>

typedef __attribute__((ext_vector_type(16))) _Float16 v16h;
typedef __attribute__((ext_vector_type(8)))  float    v8f;

#define IMG_H  512
#define IMG_W  512
#define NB     8
#define KTAPS  289          // 17*17
#define NCHUNK 10           // K padded to 320 = 10 x 32
// A-fragment store: [tile t(2)][chunk c(10)][lane(32)][j(16)] halves
#define AFRAG_HALVES (2 * NCHUNK * 32 * 16)

// ---------------------------------------------------------------------------
// Kernel 1: build the 32-orientation DoG bank directly in the WMMA A-fragment
// VGPR layout (f16).  16x32 f16 A tile, this code's contiguous k order:
//   lanes 0-15 : M = lane,    K=j
//   lanes16-31 : M = lane-16, K=j+16
// ---------------------------------------------------------------------------
__global__ __launch_bounds__(256) void Orient_build_afrag(_Float16* __restrict__ afrag) {
    const float PI = 3.14159265358979323846f;
    for (int i = threadIdx.x; i < AFRAG_HALVES; i += 256) {
        int j    = i & 15;
        int lane = (i >> 4) & 31;
        int rem  = i >> 9;          // (t*10 + c)
        int c    = rem % NCHUNK;
        int t    = rem / NCHUNK;

        int m = (lane & 15) + 16 * t;                       // channel 0..31
        int klocal = j + ((lane & 16) ? 16 : 0);
        int k = c * 32 + klocal;                            // tap index

        float val = 0.0f;
        if (k < KTAPS) {
            int r  = k / 17;        // row (x coordinate in reference)
            int cc = k - r * 17;    // col (y coordinate)
            float fx = (float)r  - 8.0f;
            float fy = (float)cc - 8.0f;
            float th = (PI / 32.0f) * (float)m;
            float co = __cosf(th), si = __sinf(th);
            float xt =  fx * co + fy * si;
            float yt = -fx * si + fy * co;
            // sigma_h=1, sigma_l=2, sigma_y=2 ; gb = (g_h - g_l)/(1 - 1/2)
            float gh = __expf(-0.5f * (xt * xt          + yt * yt * 0.25f));
            float gl = __expf(-0.5f * (xt * xt * 0.25f  + yt * yt * 0.25f)) * 0.5f;
            val = 2.0f * (gh - gl);
        }
        afrag[i] = (_Float16)val;
    }
}

// ---------------------------------------------------------------------------
// Kernel 2: implicit-GEMM conv + ReLU + channel argmax/max.
// Block = 256 threads = 8 waves; block tile = 16(x) x 16(y) pixels.
// Each wave computes TWO rows of 16 pixels (rows wave and wave+8) for all 32
// channels: 4 accumulators x 10 K-chunks of v_wmma_f32_16x16x32_f16, with
// each A-fragment load shared by both rows.  B fragments are gathered from
// an f16 LDS tile with compile-time DS offset immediates.
// ---------------------------------------------------------------------------
__global__ __launch_bounds__(256) void Orient_conv_wmma(const float* __restrict__ img,
                                                        const _Float16* __restrict__ afrag,
                                                        float* __restrict__ out) {
    __shared__ _Float16 ldsImg[32 * 32];   // f16 tile + halo, 2 KB

    const int tid  = threadIdx.x;
    const int wave = tid >> 5;
    const int lane = tid & 31;
    const int x0   = blockIdx.x * 16;
    const int y0   = blockIdx.y * 16;
    const int b    = blockIdx.z;

    // Stage 32x32 image tile (origin y0-8, x0-8), zero-padded, f32 -> f16.
    {
        const float* imgb = img + (size_t)b * IMG_H * IMG_W;
        #pragma unroll
        for (int i = 0; i < 4; ++i) {
            int idx = tid + i * 256;
            int row = idx >> 5, col = idx & 31;
            int iy = y0 - 8 + row, ix = x0 - 8 + col;
            float v = 0.0f;
            if (iy >= 0 && iy < IMG_H && ix >= 0 && ix < IMG_W)
                v = imgb[iy * IMG_W + ix];
            ldsImg[idx] = (_Float16)v;
        }
    }
    __syncthreads();

    const int n   = lane & 15;     // pixel within 16-wide strip
    const int grp = lane >> 4;     // B layout: lanes 16-31 hold K+16
    const int baseA = wave * 32 + n;          // row  wave
    const int baseB = (wave + 8) * 32 + n;    // row  wave+8

    // off(k+16)-off(k) == 31 except when k%17==0 (then 16): two base ptrs per
    // row cover every tap with a compile-time offset immediate.
    const _Float16* pA31 = &ldsImg[baseA + grp * 31];
    const _Float16* pA16 = &ldsImg[baseA + grp * 16];
    const _Float16* pB31 = &ldsImg[baseB + grp * 31];
    const _Float16* pB16 = &ldsImg[baseB + grp * 16];

    v8f accA0 = {}, accA1 = {};    // row A: channels 0..15 / 16..31
    v8f accB0 = {}, accB1 = {};    // row B: channels 0..15 / 16..31

    #pragma unroll
    for (int c = 0; c < 9; ++c) {
        v16h a0 = *(const v16h*)&afrag[(c * 32 + lane) * 16];
        v16h a1 = *(const v16h*)&afrag[((NCHUNK + c) * 32 + lane) * 16];

        union { _Float16 h[16]; v16h v; } bA, bB;
        #pragma unroll
        for (int j = 0; j < 16; ++j) {
            const int k0  = c * 32 + j;                    // grp0 tap (const)
            const int off = (k0 / 17) * 32 + (k0 % 17);    // const immediate
            if ((k0 % 17) == 0) { bA.h[j] = pA16[off]; bB.h[j] = pB16[off]; }
            else                { bA.h[j] = pA31[off]; bB.h[j] = pB31[off]; }
        }

        accA0 = __builtin_amdgcn_wmma_f32_16x16x32_f16(
            false, a0, false, bA.v, (short)0, accA0, false, false);
        accA1 = __builtin_amdgcn_wmma_f32_16x16x32_f16(
            false, a1, false, bA.v, (short)0, accA1, false, false);
        accB0 = __builtin_amdgcn_wmma_f32_16x16x32_f16(
            false, a0, false, bB.v, (short)0, accB0, false, false);
        accB1 = __builtin_amdgcn_wmma_f32_16x16x32_f16(
            false, a1, false, bB.v, (short)0, accB1, false, false);
    }

    // Chunk 9: only tap k=288 (grp0, j=0) is inside the 289-tap kernel.
    {
        v16h a0 = *(const v16h*)&afrag[(9 * 32 + lane) * 16];
        v16h a1 = *(const v16h*)&afrag[(19 * 32 + lane) * 16];

        union { _Float16 h[16]; v16h v; } bA, bB;
        #pragma unroll
        for (int j = 0; j < 16; ++j) { bA.h[j] = (_Float16)0.0f; bB.h[j] = (_Float16)0.0f; }
        const int off288 = (288 / 17) * 32 + (288 % 17);   // 528
        _Float16 vA = ldsImg[baseA + off288];
        _Float16 vB = ldsImg[baseB + off288];
        bA.h[0] = grp ? (_Float16)0.0f : vA;
        bB.h[0] = grp ? (_Float16)0.0f : vB;

        accA0 = __builtin_amdgcn_wmma_f32_16x16x32_f16(
            false, a0, false, bA.v, (short)0, accA0, false, false);
        accA1 = __builtin_amdgcn_wmma_f32_16x16x32_f16(
            false, a1, false, bA.v, (short)0, accA1, false, false);
        accB0 = __builtin_amdgcn_wmma_f32_16x16x32_f16(
            false, a0, false, bB.v, (short)0, accB0, false, false);
        accB1 = __builtin_amdgcn_wmma_f32_16x16x32_f16(
            false, a1, false, bB.v, (short)0, accB1, false, false);
    }

    // ReLU + argmax/max. D layout: lanes 0-15 -> M=v, lanes 16-31 -> M=8+v.
    const size_t BHW = (size_t)NB * IMG_H * IMG_W;

    #pragma unroll
    for (int row = 0; row < 2; ++row) {
        const v8f& c0 = row ? accB0 : accA0;
        const v8f& c1 = row ? accB1 : accA1;

        float best = -1.0f;
        int   bm   = 0;
        #pragma unroll
        for (int v = 0; v < 8; ++v) {
            float r0 = fmaxf(c0[v], 0.0f);
            int   m0 = grp * 8 + v;
            if (r0 > best) { best = r0; bm = m0; }
        }
        #pragma unroll
        for (int v = 0; v < 8; ++v) {
            float r1 = fmaxf(c1[v], 0.0f);
            int   m1 = 16 + grp * 8 + v;
            if (r1 > best) { best = r1; bm = m1; }
        }

        // Combine with partner lane (xor 16); first-index tie-break (argmax).
        float obest = __shfl_xor(best, 16, 32);
        int   obm   = __shfl_xor(bm,   16, 32);
        if (obest > best || (obest == best && obm < bm)) { best = obest; bm = obm; }

        if (lane < 16) {
            int y = y0 + wave + row * 8;
            size_t idx = (size_t)b * IMG_H * IMG_W + (size_t)y * IMG_W + (x0 + n);
            out[idx]       = (float)bm;   // max_res    [B,H,W]
            out[BHW + idx] = best;        // confidence [B,1,H,W]
        }
    }
}

extern "C" void kernel_launch(void* const* d_in, const int* in_sizes, int n_in,
                              void* d_out, int out_size, void* d_ws, size_t ws_size,
                              hipStream_t stream) {
    const float* img = (const float*)d_in[0];
    float* out = (float*)d_out;
    _Float16* afrag = (_Float16*)d_ws;   // 20 KB scratch for filter fragments

    Orient_build_afrag<<<1, 256, 0, stream>>>(afrag);

    dim3 grid(IMG_W / 16, IMG_H / 16, NB);
    Orient_conv_wmma<<<grid, 256, 0, stream>>>(img, afrag, out);
}
